// InstanceBranch_51118700757225
// MI455X (gfx1250) — matrix operand order repaired and checked by reading, not compiled, not run on hardware
//
#include <hip/hip_runtime.h>
#include <stdint.h>
#include <math.h>

// ---------------- problem constants ----------------
#define B_     64
#define CIN_   256
#define H_     40
#define W_     100
#define HW_    4000      // H_*W_
#define DIM_   134
#define CPD_   160       // DIM padded to multiple of 32 (5 K-chunks)
#define NMASK_ 4
#define KDIM_  128
#define NCLS_  2

#define MTILE  32        // spatial positions per block
#define SL     36        // strip length (need 34: 32 outputs + dx halo)
#define NROWS  (3 * SL)  // 3 strips (one per dy) per buffer = 108 rows
#define ZROW   (2 * NROWS) // zero row index (shared by both buffers)

// ---------------- CDNA5 feature detection ----------------
#if defined(__has_builtin)
#  if __has_builtin(__builtin_amdgcn_global_load_async_to_lds_b128)
#    define USE_ASYNC 1
#  endif
#endif
#ifndef USE_ASYNC
#  define USE_ASYNC 0
#endif

typedef int v4i_async __attribute__((vector_size(16)));

#if USE_ASYNC
#  define TO_GLOBAL(p) ((__attribute__((address_space(1))) v4i_async*)(unsigned long long)(p))
#  define TO_LDS(p)    ((__attribute__((address_space(3))) v4i_async*)(unsigned int)(unsigned long long)(p))
#  if defined(__has_builtin) && __has_builtin(__builtin_amdgcn_s_wait_asynccnt)
#    define WAIT_ASYNC() __builtin_amdgcn_s_wait_asynccnt(0)
#  else
#    define WAIT_ASYNC() asm volatile("s_wait_asynccnt 0" ::: "memory")
#  endif
#else
#  define WAIT_ASYNC() do {} while (0)
#endif

typedef __attribute__((ext_vector_type(16))) __bf16 bf16x16;
typedef __attribute__((ext_vector_type(8)))  float  f32x8;

union FragAB { bf16x16 v; uint4 q[2]; };

__device__ __forceinline__ uint16_t f2bf(float f) {
  uint32_t u = __float_as_uint(f);
  u = u + 0x7FFFu + ((u >> 16) & 1u);           // round-to-nearest-even
  return (uint16_t)(u >> 16);
}
__device__ __forceinline__ float bf2f(uint16_t h) {
  return __uint_as_float(((uint32_t)h) << 16);
}

// ---------------- weight packing ----------------
// dst[((t*KC+kc)*NT+nt)*512 + n*32 + k] = bf16(w[co=nt*16+n][ci=kc*32+k][t])
__global__ void pack_w(const float* __restrict__ w, int Co, int Ci, int KC, int NT,
                       uint16_t* __restrict__ dst) {
  int total = 9 * KC * NT * 512;
  for (int idx = blockIdx.x * blockDim.x + threadIdx.x; idx < total;
       idx += gridDim.x * blockDim.x) {
    int k    = idx & 31;
    int n    = (idx >> 5) & 15;
    int rest = idx >> 9;
    int nt   = rest % NT;
    int rest2 = rest / NT;
    int kc   = rest2 % KC;
    int t    = rest2 / KC;
    int co = nt * 16 + n, ci = kc * 32 + k;
    float v = (co < Co && ci < Ci) ? w[((size_t)co * Ci + ci) * 9 + t] : 0.f;
    dst[idx] = f2bf(v);
  }
}

__global__ void pack_bias(const float* __restrict__ b, int Co, int Cp,
                          float* __restrict__ dst) {
  int i = blockIdx.x * blockDim.x + threadIdx.x;
  if (i < Cp) dst[i] = (i < Co) ? b[i] : 0.f;
}

// ---------------- implicit-GEMM 3x3 SAME conv via WMMA bf16 ----------------
// grid = (HW/32, B), block = 320 (10 waves; wave w owns output channels [16w,16w+16))
// LDS: double-buffered, 3 halo strips (dy=0,1,2) of 36 rows x 32 bf16 channels.
__global__ __launch_bounds__(320)
void conv3x3_wmma(const float* __restrict__ srcF,    // mode 0: fp32 NCHW
                  const uint16_t* __restrict__ srcH, // mode 1: bf16 NHWC (stride CPD_)
                  int srcMode, int Cin, int KC, int NT,
                  const uint16_t* __restrict__ wPack,
                  const float* __restrict__ bias,
                  uint16_t* __restrict__ dstFeat,    // bf16 NHWC out (non-iam)
                  int iamMode,
                  float* __restrict__ iamOut,        // raw logits, NCHW
                  float* __restrict__ iamProb) {     // sigmoid, NCHW
  __shared__ __align__(16) uint16_t stage[(2 * NROWS + 1) * 32];
  const int tid  = threadIdx.x;
  const int lane = tid & 31;
  const int wave = tid >> 5;
  const int b    = blockIdx.y;
  const int p0   = blockIdx.x * MTILE;

  if (tid < 32) stage[ZROW * 32 + tid] = 0;   // zero row for SAME-pad columns

  // stage one 32-channel K-chunk into buffer `buf` (async on CDNA5 for bf16 src)
  auto stageChunk = [&](int buf, int kc) {
    const int base = buf * NROWS;
    if (srcMode == 0) {
      // fp32 NCHW source: convert on the fly (layer 0 only)
      for (int d = tid; d < NROWS * 16; d += 320) {   // dword (2 x bf16) granularity
        int li = d >> 4, pr = d & 15;
        int strip = li / SL, r = li - strip * SL;
        int q = p0 + (strip - 1) * W_ - 1 + r;
        uint32_t val = 0;
        if (q >= 0 && q < HW_) {
          int c = kc * 32 + pr * 2;
          float f0 = srcF[((size_t)b * Cin + c) * HW_ + q];
          float f1 = srcF[((size_t)b * Cin + c + 1) * HW_ + q];
          val = (uint32_t)f2bf(f0) | ((uint32_t)f2bf(f1) << 16);
        }
        *(uint32_t*)&stage[(base + li) * 32 + pr * 2] = val;
      }
    } else {
      for (int d = tid; d < NROWS * 4; d += 320) {    // 16B granularity
        int li = d >> 2, part = d & 3;
        int strip = li / SL, r = li - strip * SL;
        int q = p0 + (strip - 1) * W_ - 1 + r;
        uint16_t* lp = &stage[(base + li) * 32 + part * 8];
        if (q >= 0 && q < HW_) {
          const uint16_t* gp = srcH + ((size_t)b * HW_ + q) * CPD_ + kc * 32 + part * 8;
#if USE_ASYNC
          __builtin_amdgcn_global_load_async_to_lds_b128(TO_GLOBAL(gp), TO_LDS(lp), 0, 0);
#else
          *(uint4*)lp = *(const uint4*)gp;
#endif
        } else {
          uint4 z = {0, 0, 0, 0};
          *(uint4*)lp = z;
        }
      }
    }
  };

  f32x8 acc0 = {};
  f32x8 acc1 = {};

  const int mrow   = lane & 15;
  const int w0c    = (p0 + mrow) % W_;        // output column, M-subtile 0
  const int w1c    = (p0 + 16 + mrow) % W_;   // output column, M-subtile 1
  const int chBase = (lane < 16) ? 0 : 8;     // A-fragment K sub-chunk per lane half

  // prologue: fill buffer 0
  stageChunk(0, 0);
  WAIT_ASYNC();
  __syncthreads();

  for (int kc = 0; kc < KC; ++kc) {
    const int buf = kc & 1;
    // prefetch next K-chunk into the other buffer while we compute
    if (kc + 1 < KC) stageChunk(buf ^ 1, kc + 1);

    if (wave < NT) {
      const int base = buf * NROWS;
      for (int t = 0; t < 9; ++t) {
        const int dy = t / 3, dx = t % 3;
        // B fragment: N=lane%16 column, K sub-chunks chBase..+7 and chBase+16..+23
        const uint16_t* wf = wPack + (((size_t)(t * KC + kc)) * NT + wave) * 512;
        FragAB fb;
        fb.q[0] = *(const uint4*)(wf + mrow * 32 + chBase);
        fb.q[1] = *(const uint4*)(wf + mrow * 32 + chBase + 16);
        // A fragments from LDS strips (zero row when SAME-pad column is OOB)
        bool v0 = (unsigned)(w0c + dx - 1) < (unsigned)W_;
        bool v1 = (unsigned)(w1c + dx - 1) < (unsigned)W_;
        int li0 = v0 ? (base + dy * SL + mrow + dx)      : ZROW;
        int li1 = v1 ? (base + dy * SL + 16 + mrow + dx) : ZROW;
        FragAB fa0, fa1;
        fa0.q[0] = *(const uint4*)&stage[li0 * 32 + chBase];
        fa0.q[1] = *(const uint4*)&stage[li0 * 32 + chBase + 16];
        fa1.q[0] = *(const uint4*)&stage[li1 * 32 + chBase];
        fa1.q[1] = *(const uint4*)&stage[li1 * 32 + chBase + 16];
        acc0 = __builtin_amdgcn_wmma_f32_16x16x32_bf16(false, fa0.v, false, fb.v,
                                                       (short)0, acc0, false, false);
        acc1 = __builtin_amdgcn_wmma_f32_16x16x32_bf16(false, fa1.v, false, fb.v,
                                                       (short)0, acc1, false, false);
      }
    }

    WAIT_ASYNC();     // next chunk landed in LDS
    __syncthreads();  // all waves done computing `buf` before it is re-staged
  }

  if (wave >= NT) return;

  const int ncol = wave * 16 + (lane & 15);
  const float bv = bias[ncol];
  for (int e = 0; e < 8; ++e) {
    int m = e + ((lane >> 4) << 3);
    float v0 = acc0[e] + bv;
    float v1 = acc1[e] + bv;
    int pe0 = p0 + m;
    int pe1 = p0 + 16 + m;
    if (!iamMode) {
      v0 = v0 > 0.f ? v0 : 0.f;
      v1 = v1 > 0.f ? v1 : 0.f;
      dstFeat[((size_t)b * HW_ + pe0) * CPD_ + ncol] = f2bf(v0);
      dstFeat[((size_t)b * HW_ + pe1) * CPD_ + ncol] = f2bf(v1);
    } else if (ncol < NMASK_) {
      size_t base = ((size_t)b * NMASK_ + ncol) * HW_;
      iamOut[base + pe0]  = v0;
      iamOut[base + pe1]  = v1;
      iamProb[base + pe0] = 1.f / (1.f + expf(-v0));
      iamProb[base + pe1] = 1.f / (1.f + expf(-v1));
    }
  }
}

// ---------------- per-(b,n) argmax + sum over spatial ----------------
__global__ __launch_bounds__(256)
void reduce_iam(const float* __restrict__ prob, int* __restrict__ pos,
                float* __restrict__ norm) {
  __shared__ float smx[256];
  __shared__ int   sidx[256];
  __shared__ float ssum[256];
  int row = blockIdx.x, tid = threadIdx.x;
  const float* pr = prob + (size_t)row * HW_;
  float mx = -3.4e38f; int mi = 0; float sm = 0.f;
  for (int i = tid; i < HW_; i += 256) {
    float v = pr[i];
    sm += v;
    if (v > mx) { mx = v; mi = i; }
  }
  smx[tid] = mx; sidx[tid] = mi; ssum[tid] = sm;
  __syncthreads();
  for (int s = 128; s > 0; s >>= 1) {
    if (tid < s) {
      ssum[tid] += ssum[tid + s];
      float om = smx[tid + s]; int oi = sidx[tid + s];
      if (om > smx[tid] || (om == smx[tid] && oi < sidx[tid])) {
        smx[tid] = om; sidx[tid] = oi;
      }
    }
    __syncthreads();
  }
  if (tid == 0) {
    pos[row] = sidx[0];
    float s = ssum[0];
    norm[row] = s < 1e-6f ? 1e-6f : s;
  }
}

// ---------------- normalized iam, padded to 16 rows, as bf16 ----------------
__global__ void build_iamn(const float* __restrict__ prob, const float* __restrict__ norm,
                           uint16_t* __restrict__ iamn) {
  int total = B_ * 16 * HW_;
  for (int idx = blockIdx.x * blockDim.x + threadIdx.x; idx < total;
       idx += gridDim.x * blockDim.x) {
    int s = idx % HW_;
    int r = (idx / HW_) & 15;
    int b = idx / (16 * HW_);
    float v = 0.f;
    if (r < NMASK_) v = prob[((size_t)b * NMASK_ + r) * HW_ + s] / norm[b * NMASK_ + r];
    iamn[idx] = f2bf(v);
  }
}

// ---------------- seed params gather ----------------
__global__ void gather_seed(const uint16_t* __restrict__ feat, const int* __restrict__ pos,
                            float* __restrict__ outMask, float* __restrict__ outReg) {
  int idx = blockIdx.x * blockDim.x + threadIdx.x;
  int total = B_ * NMASK_ * DIM_;
  if (idx >= total) return;
  int c = idx % DIM_, row = idx / DIM_;
  int b = row / NMASK_;
  int p = pos[row];
  float v = bf2f(feat[((size_t)b * HW_ + p) * CPD_ + c]);
  if (c < 67) outMask[row * 67 + c] = v;
  else        outReg[row * 67 + (c - 67)] = v;
}

// ---------------- inst = iam_n @ feat  (M=16 pad, K=4000, N=160) via WMMA ----------------
__global__ __launch_bounds__(320)
void einsum_wmma(const uint16_t* __restrict__ iamn, const uint16_t* __restrict__ feat,
                 float* __restrict__ inst) {
  const int tid = threadIdx.x, lane = tid & 31, wave = tid >> 5;
  const int b = blockIdx.x;
  f32x8 acc = {};
  const int mrow   = lane & 15;
  const int chBase = (lane < 16) ? 0 : 8;
  const uint16_t* An = iamn + (size_t)b * 16 * HW_ + (size_t)mrow * HW_;
  const uint16_t* Bf = feat + (size_t)b * HW_ * CPD_ + wave * 16;
  for (int sc = 0; sc < HW_ / 32; ++sc) {
    FragAB fa, fb;
    int s0 = sc * 32;
    fa.q[0] = *(const uint4*)(An + s0 + chBase);
    fa.q[1] = *(const uint4*)(An + s0 + chBase + 16);
    const uint16_t* bp = Bf + (size_t)(s0 + chBase) * CPD_ + mrow;
    for (int s = 0; s < 16; ++s)
      ((uint16_t*)&fb)[s] = bp[(size_t)(s + (s >= 8 ? 8 : 0)) * CPD_];
    acc = __builtin_amdgcn_wmma_f32_16x16x32_bf16(false, fa.v, false, fb.v,
                                                  (short)0, acc, false, false);
  }
  const int ncol = wave * 16 + (lane & 15);
  for (int e = 0; e < 8; ++e) {
    int m = e + ((lane >> 4) << 3);
    inst[((size_t)b * 16 + m) * CPD_ + ncol] = acc[e];
  }
}

// ---------------- classification + kernel heads ----------------
__global__ void heads(const float* __restrict__ inst,
                      const float* __restrict__ cls_w, const float* __restrict__ cls_b,
                      const float* __restrict__ mk_w,  const float* __restrict__ mk_b,
                      float* __restrict__ outLogits, float* __restrict__ outKernel) {
  int idx = blockIdx.x * blockDim.x + threadIdx.x;
  int total = B_ * NMASK_ * (NCLS_ + KDIM_);
  if (idx >= total) return;
  int o = idx % (NCLS_ + KDIM_), row = idx / (NCLS_ + KDIM_);
  int b = row / NMASK_, n = row % NMASK_;
  const float* iv = inst + ((size_t)b * 16 + n) * CPD_;
  if (o < NCLS_) {
    float s = cls_b[o];
    for (int c = 0; c < DIM_; ++c) s += iv[c] * cls_w[o * DIM_ + c];
    outLogits[row * NCLS_ + o] = s;
  } else {
    int k = o - NCLS_;
    float s = mk_b[k];
    for (int c = 0; c < DIM_; ++c) s += iv[c] * mk_w[k * DIM_ + c];
    outKernel[(size_t)row * KDIM_ + k] = s;
  }
}

// ---------------- host-side orchestration ----------------
extern "C" void kernel_launch(void* const* d_in, const int* in_sizes, int n_in,
                              void* d_out, int out_size, void* d_ws, size_t ws_size,
                              hipStream_t stream) {
  (void)in_sizes; (void)n_in; (void)out_size; (void)ws_size;
  const float* x     = (const float*)d_in[0];
  const float* w0    = (const float*)d_in[1];
  const float* b0    = (const float*)d_in[2];
  const float* w1    = (const float*)d_in[3];
  const float* b1    = (const float*)d_in[4];
  const float* w2    = (const float*)d_in[5];
  const float* b2    = (const float*)d_in[6];
  const float* w3    = (const float*)d_in[7];
  const float* b3    = (const float*)d_in[8];
  const float* iam_w = (const float*)d_in[9];
  const float* iam_b = (const float*)d_in[10];
  const float* cls_w = (const float*)d_in[11];
  const float* cls_b = (const float*)d_in[12];
  const float* mk_w  = (const float*)d_in[13];
  const float* mk_b  = (const float*)d_in[14];
  float* out = (float*)d_out;

  // output offsets (flat, return order)
  const size_t o_logits = 0;
  const size_t o_mask   = o_logits + (size_t)B_ * NMASK_ * NCLS_;   // 512
  const size_t o_reg    = o_mask + (size_t)B_ * NMASK_ * 67;        // +17152
  const size_t o_pk     = o_reg + (size_t)B_ * NMASK_ * 67;         // +17152
  const size_t o_iam    = o_pk + (size_t)B_ * NMASK_ * KDIM_;       // +32768

  // workspace carve-out
  char* ws = (char*)d_ws;
  size_t off = 0;
  auto alloc = [&](size_t bytes) -> char* {
    char* p = ws + off;
    off += (bytes + 255) & ~(size_t)255;
    return p;
  };
  uint16_t* featA = (uint16_t*)alloc((size_t)B_ * HW_ * CPD_ * 2);
  uint16_t* featB = (uint16_t*)alloc((size_t)B_ * HW_ * CPD_ * 2);
  uint16_t* w0p = (uint16_t*)alloc((size_t)9 * 8 * 10 * 512 * 2);
  uint16_t* w1p = (uint16_t*)alloc((size_t)9 * 5 * 10 * 512 * 2);
  uint16_t* w2p = (uint16_t*)alloc((size_t)9 * 5 * 10 * 512 * 2);
  uint16_t* w3p = (uint16_t*)alloc((size_t)9 * 5 * 10 * 512 * 2);
  uint16_t* wip = (uint16_t*)alloc((size_t)9 * 5 * 1 * 512 * 2);
  float* b0p = (float*)alloc(CPD_ * 4);
  float* b1p = (float*)alloc(CPD_ * 4);
  float* b2p = (float*)alloc(CPD_ * 4);
  float* b3p = (float*)alloc(CPD_ * 4);
  float* bip = (float*)alloc(16 * 4);
  float*    iamProb = (float*)alloc((size_t)B_ * NMASK_ * HW_ * 4);
  uint16_t* iamn    = (uint16_t*)alloc((size_t)B_ * 16 * HW_ * 2);
  int*      pos     = (int*)alloc(B_ * NMASK_ * 4);
  float*    norm    = (float*)alloc(B_ * NMASK_ * 4);
  float*    inst    = (float*)alloc((size_t)B_ * 16 * CPD_ * 4);

  // weight/bias packing (deterministic, every call)
  pack_w<<<dim3(512), 256, 0, stream>>>(w0, DIM_, CIN_, 8, 10, w0p);
  pack_w<<<dim3(512), 256, 0, stream>>>(w1, DIM_, DIM_, 5, 10, w1p);
  pack_w<<<dim3(512), 256, 0, stream>>>(w2, DIM_, DIM_, 5, 10, w2p);
  pack_w<<<dim3(512), 256, 0, stream>>>(w3, DIM_, DIM_, 5, 10, w3p);
  pack_w<<<dim3(64),  256, 0, stream>>>(iam_w, NMASK_, DIM_, 5, 1, wip);
  pack_bias<<<1, 256, 0, stream>>>(b0, DIM_, CPD_, b0p);
  pack_bias<<<1, 256, 0, stream>>>(b1, DIM_, CPD_, b1p);
  pack_bias<<<1, 256, 0, stream>>>(b2, DIM_, CPD_, b2p);
  pack_bias<<<1, 256, 0, stream>>>(b3, DIM_, CPD_, b3p);
  pack_bias<<<1, 256, 0, stream>>>(iam_b, NMASK_, 16, bip);

  // conv tower (ping-pong bf16 NHWC feature buffers)
  dim3 cgrid(HW_ / MTILE, B_);
  conv3x3_wmma<<<cgrid, 320, 0, stream>>>(x, nullptr, 0, CIN_, 8, 10, w0p, b0p,
                                          featA, 0, nullptr, nullptr);
  conv3x3_wmma<<<cgrid, 320, 0, stream>>>(nullptr, featA, 1, DIM_, 5, 10, w1p, b1p,
                                          featB, 0, nullptr, nullptr);
  conv3x3_wmma<<<cgrid, 320, 0, stream>>>(nullptr, featB, 1, DIM_, 5, 10, w2p, b2p,
                                          featA, 0, nullptr, nullptr);
  conv3x3_wmma<<<cgrid, 320, 0, stream>>>(nullptr, featA, 1, DIM_, 5, 10, w3p, b3p,
                                          featB, 0, nullptr, nullptr);
  // iam conv: raw logits straight to output, sigmoid probs to workspace
  conv3x3_wmma<<<cgrid, 320, 0, stream>>>(nullptr, featB, 1, DIM_, 5, 1, wip, bip,
                                          nullptr, 1, out + o_iam, iamProb);

  reduce_iam<<<B_ * NMASK_, 256, 0, stream>>>(iamProb, pos, norm);
  build_iamn<<<dim3(2048), 256, 0, stream>>>(iamProb, norm, iamn);
  gather_seed<<<(B_ * NMASK_ * DIM_ + 255) / 256, 256, 0, stream>>>(
      featB, pos, out + o_mask, out + o_reg);
  einsum_wmma<<<B_, 320, 0, stream>>>(iamn, featB, inst);
  heads<<<(B_ * NMASK_ * (NCLS_ + KDIM_) + 255) / 256, 256, 0, stream>>>(
      inst, cls_w, cls_b, mk_w, mk_b, out + o_logits, out + o_pk);
}
